// DepthFormer_71485435675050
// MI455X (gfx1250) — compile-verified, matrix operand order checked by reading, not run on hardware
//
#include <hip/hip_runtime.h>
#include <hip/hip_bf16.h>

// ---------------------------------------------------------------------------
// DepthFormer for MI455X (gfx1250, wave32, WMMA).
// All GEMM operands live in f16 workspace -> fragments load as 2x b128 per
// lane with zero conversion VALU; accumulation f32 in v_wmma_f32_16x16x32_f16.
// Residual stream / softmax logits / norms stay f32. Each wave computes an
// (16*MT)x(16*NT) C block for operand reuse. Peak workspace ~250 MB.
// ---------------------------------------------------------------------------

typedef __attribute__((ext_vector_type(16))) _Float16 v16h;
typedef __attribute__((ext_vector_type(8)))  _Float16 v8h;
typedef __attribute__((ext_vector_type(8)))  float    v8f;
typedef _Float16 f16;

#define DEV static __device__ __forceinline__

// ---- f16 WMMA fragment loaders (CDNA5 ISA 7.12.2, wave32) ------------------
// 16-bit A 16x32: lane<16 -> row lane, K {0..7,16..23}; lane>=16 -> K +8.
// Element e of v16h maps to K = half*8 + (e<8 ? e : e+8): two contiguous
// 8-halfword (16B) runs -> two b128 loads per lane.
DEV v16h frag_A(const f16* __restrict__ A, int lda) {
  const int lane = threadIdx.x & 31;
  const int half = lane >> 4;
  const int r    = lane & 15;
  const f16* p = A + (size_t)r * lda + half * 8;
  v8h lo = *(const v8h*)p;
  v8h hi = *(const v8h*)(p + 16);
  v16h a;
#pragma unroll
  for (int i = 0; i < 8; ++i) { a[i] = lo[i]; a[8 + i] = hi[i]; }
  return a;
}

// B = Bt^T, Bt row-major [N,K]; tile origin passed as Bt + n0*ldbt + k0.
DEV v16h frag_Bt(const f16* __restrict__ Bt, int ldbt) {
  const int lane = threadIdx.x & 31;
  const int half = lane >> 4;
  const int c    = lane & 15;
  const f16* p = Bt + (size_t)c * ldbt + half * 8;
  v8h lo = *(const v8h*)p;
  v8h hi = *(const v8h*)(p + 16);
  v16h b;
#pragma unroll
  for (int i = 0; i < 8; ++i) { b[i] = lo[i]; b[8 + i] = hi[i]; }
  return b;
}

// B row-major [K,N] tile (k0,n0): strided 16-bit gathers (used where no
// transposed copy exists).
DEV v16h frag_B(const f16* __restrict__ B, int ldb) {
  const int lane = threadIdx.x & 31;
  const int half = lane >> 4;
  const int c    = lane & 15;
  v16h b;
#pragma unroll
  for (int v = 0; v < 8; ++v) {
    const int k = ((v < 4) ? 2 * v : 2 * v + 8) + half * 8;
    b[2 * v]     = B[(size_t)k * ldb + c];
    b[2 * v + 1] = B[(size_t)(k + 1) * ldb + c];
  }
  return b;
}

// ---- Batched WMMA GEMM: C = alpha*A@B (+bias f32) (+res f32) ---------------
// One wave/block computing (16*MT)x(16*NT) of C. K multiple of 32.
template <int MT, int NT, bool TB, typename TO>
__global__ __launch_bounds__(32) void k_gemm(
    const f16* __restrict__ A, const f16* __restrict__ B, TO* __restrict__ C,
    int lda, int ldb, int ldc, int K, long long sA, long long sB, long long sC,
    float alpha, const float* __restrict__ bias,
    const float* __restrict__ res) {
  const int m0 = blockIdx.x * (16 * MT);
  const int n0 = blockIdx.y * (16 * NT);
  const long long bz = blockIdx.z;
  A += bz * sA;
  B += bz * sB;
  C += bz * sC;
  const float* R = res ? res + bz * sC : nullptr;

  v8f acc[MT][NT] = {};
  for (int k0 = 0; k0 < K; k0 += 32) {
    v16h af[MT];
#pragma unroll
    for (int i = 0; i < MT; ++i) {
      const f16* Ap = A + (size_t)(m0 + 16 * i) * lda + k0;
      if (k0 + 32 < K) __builtin_prefetch(Ap + 32, 0, 1);  // global_prefetch_b8
      af[i] = frag_A(Ap, lda);
    }
#pragma unroll
    for (int j = 0; j < NT; ++j) {
      v16h bf;
      if (TB) bf = frag_Bt(B + (size_t)(n0 + 16 * j) * ldb + k0, ldb);
      else    bf = frag_B (B + (size_t)k0 * ldb + (n0 + 16 * j), ldb);
#pragma unroll
      for (int i = 0; i < MT; ++i)
        acc[i][j] = __builtin_amdgcn_wmma_f32_16x16x32_f16(
            false, af[i], false, bf, (short)0, acc[i][j], false, false);
    }
  }
  // C/D f32 tile layout: lane&15 = col, VGPR v -> row v (+8 for lanes 16-31).
  const int lane = threadIdx.x & 31;
  const int half = lane >> 4;
  const int c    = lane & 15;
#pragma unroll
  for (int i = 0; i < MT; ++i)
#pragma unroll
    for (int j = 0; j < NT; ++j) {
      const int n = n0 + 16 * j + c;
#pragma unroll
      for (int v = 0; v < 8; ++v) {
        const int m = m0 + 16 * i + v + half * 8;
        float val = acc[i][j][v] * alpha;
        if (bias) val += bias[n];
        if (R) val += R[(size_t)m * ldc + n];
        C[(size_t)m * ldc + n] = (TO)val;
      }
    }
}

// ---- Elementwise / reduction kernels ---------------------------------------

// ScaleNorm row(192): y = x / max(||x||,1e-5) * g[gi]; f32 in, TO out.
template <typename TO>
__global__ __launch_bounds__(128) void k_scale_norm(
    const float* __restrict__ X, TO* __restrict__ Y,
    const float* __restrict__ g, int gi) {
  const long long row = (long long)blockIdx.x * 4 + (threadIdx.x >> 5);
  const int lane = threadIdx.x & 31;
  const float* p = X + row * 192;
  float s = 0.f;
  for (int j = lane; j < 192; j += 32) { float v = p[j]; s += v * v; }
  for (int o = 16; o > 0; o >>= 1) s += __shfl_down(s, o, 32);
  s = __shfl(s, 0, 32);
  const float sc = g[gi] / fmaxf(sqrtf(s), 1e-5f);
  for (int j = lane; j < 192; j += 32) Y[row * 192 + j] = (TO)(p[j] * sc);
}

// Row softmax: f32 logits in, f16 probs out. One wave per row.
__global__ __launch_bounds__(128) void k_softmax(const float* __restrict__ X,
                                                 f16* __restrict__ O,
                                                 int ncols) {
  const long long row = (long long)blockIdx.x * 4 + (threadIdx.x >> 5);
  const int lane = threadIdx.x & 31;
  const float* p = X + row * (long long)ncols;
  f16* q = O + row * (long long)ncols;
  float mx = -3.0e38f;
  for (int j = lane; j < ncols; j += 32) mx = fmaxf(mx, p[j]);
  for (int o = 16; o > 0; o >>= 1) mx = fmaxf(mx, __shfl_down(mx, o, 32));
  mx = __shfl(mx, 0, 32);
  float s = 0.f;
  for (int j = lane; j < ncols; j += 32) s += expf(p[j] - mx);
  for (int o = 16; o > 0; o >>= 1) s += __shfl_down(s, o, 32);
  s = __shfl(s, 0, 32);
  const float inv = 1.f / s;
  for (int j = lane; j < ncols; j += 32)
    q[j] = (f16)(expf(p[j] - mx) * inv);
}

// Global max of row sums (stats[0]) / col sums (stats[1]) over batched RxC f16.
__global__ void k_rowcol_max(const f16* __restrict__ A, int R, int C,
                             long long bstride, int nB,
                             unsigned* __restrict__ stats) {
  const long long idx = (long long)blockIdx.x * blockDim.x + threadIdx.x;
  const long long nRows = (long long)nB * R;
  const long long nCols = (long long)nB * C;
  if (idx < nRows) {
    const f16* p = A + (idx / R) * bstride + (idx % R) * (long long)C;
    float s = 0.f;
    for (int j = 0; j < C; ++j) s += fabsf((float)p[j]);
    atomicMax(&stats[0], __float_as_uint(s));
  } else if (idx < nRows + nCols) {
    const long long k = idx - nRows;
    const f16* p = A + (k / C) * bstride + (k % C);
    float s = 0.f;
    for (int i = 0; i < R; ++i) s += fabsf((float)p[(long long)i * C]);
    atomicMax(&stats[1], __float_as_uint(s));
  }
}

// Z = X^T / (stats[0]*stats[1]), batched RxR, f16.
__global__ void k_pinv_seed(const f16* __restrict__ X, f16* __restrict__ Z,
                            int R, const unsigned* __restrict__ stats,
                            long long n) {
  const long long idx = (long long)blockIdx.x * blockDim.x + threadIdx.x;
  if (idx >= n) return;
  const float inv =
      1.f / (__uint_as_float(stats[0]) * __uint_as_float(stats[1]));
  const long long RR = (long long)R * R;
  const long long b = idx / RR;
  const long long r = idx % RR;
  const int i = (int)(r / R), j = (int)(r % R);
  Z[idx] = (f16)((float)X[b * RR + (long long)j * R + i] * inv);
}

// T = a*I - M, batched RxR, f16.
__global__ void k_aI_minus(float a, const f16* __restrict__ M,
                           f16* __restrict__ T, int R, long long n) {
  const long long idx = (long long)blockIdx.x * blockDim.x + threadIdx.x;
  if (idx >= n) return;
  const int i = (int)((idx / R) % R);
  const int j = (int)(idx % R);
  T[idx] = (f16)(((i == j) ? a : 0.f) - (float)M[idx]);
}

// Nystrom landmarks: mean over 4 consecutive tokens (spatial fold), f16 QKV.
__global__ void k_landmarks(const f16* __restrict__ QKV, f16* __restrict__ qL,
                            f16* __restrict__ kL, long long n) {
  const long long idx = (long long)blockIdx.x * blockDim.x + threadIdx.x;
  if (idx >= n) return;
  const int d = (int)(idx % 192);
  const long long t = idx / 192;
  const int j = (int)(t % 256);
  const int f = (int)(t / 256);
  const f16* base = QKV + ((long long)f * 1024 + (long long)j * 4) * 576;
  float q = 0.f, k = 0.f;
#pragma unroll
  for (int u = 0; u < 4; ++u) {
    q += (float)base[u * 576 + d];
    k += (float)base[u * 576 + 192 + d];
  }
  qL[idx] = (f16)(q * 0.25f);
  kL[idx] = (f16)(k * 0.25f);
}

// GEGLU (exact erf gelu): G[s,j] = H[s,j] * gelu(H[s,768+j]); f16 in/out.
__global__ void k_geglu(const f16* __restrict__ H, f16* __restrict__ G,
                        long long n) {
  const long long idx = (long long)blockIdx.x * blockDim.x + threadIdx.x;
  if (idx >= n) return;
  const long long r = idx / 768;
  const int j = (int)(idx % 768);
  const float a = (float)H[r * 1536 + j];
  const float x = (float)H[r * 1536 + 768 + j];
  G[idx] = (f16)(a * 0.5f * x * (1.f + erff(x * 0.7071067811865475f)));
}

// volume (1,3,32,256,256) -> f16 patches [32768, 192], feature (p1 p2 c).
__global__ void k_patchify(const float* __restrict__ vol, f16* __restrict__ P,
                           long long n) {
  const long long idx = (long long)blockIdx.x * blockDim.x + threadIdx.x;
  if (idx >= n) return;
  const int f = (int)(idx % 192);
  const long long t = idx / 192;
  const int w = (int)(t % 32);
  const int h = (int)((t / 32) % 32);
  const int d = (int)(t / 1024);
  const int c = f % 3;
  const int p2 = (f / 3) % 8;
  const int p1 = f / 24;
  P[idx] =
      (f16)vol[(((long long)c * 32 + d) * 256 + h * 8 + p1) * 256 + w * 8 + p2];
}

// W [K,N] f32 -> Wt [N,K] f16 (convert + transpose once per call).
__global__ void k_cvt_t(const float* __restrict__ W, f16* __restrict__ Wt,
                        int Kd, int Nd, long long n) {
  const long long idx = (long long)blockIdx.x * blockDim.x + threadIdx.x;
  if (idx >= n) return;
  const int k = (int)(idx / Nd);
  const int nn = (int)(idx % Nd);
  Wt[(size_t)nn * Kd + k] = (f16)W[idx];
}

// ---- Host helpers ----------------------------------------------------------

static inline dim3 ewgrid(long long n) {
  return dim3((unsigned)((n + 255) / 256));
}

template <int MT, int NT, bool TB, typename TO>
static void lgemm(hipStream_t st, const f16* A, const f16* B, TO* C, int M,
                  int N, int K, int lda, int ldb, int ldc, long long sA,
                  long long sB, long long sC, int nb, float alpha,
                  const float* bias, const float* res) {
  dim3 g(M / (16 * MT), N / (16 * NT), nb), blk(32);
  k_gemm<MT, NT, TB, TO><<<g, blk, 0, st>>>(A, B, C, lda, ldb, ldc, K, sA, sB,
                                            sC, alpha, bias, res);
}

// Batched Newton-Schulz pinv (6 iters), all f16 operands, f32 accumulate.
template <int MT, int NT>
static void pinv_iters(hipStream_t st, const f16* Xs, f16** Zio, f16* W,
                       f16* T, f16* U, int R, int nb) {
  const long long RR = (long long)R * R;
  const long long n = RR * nb;
  f16* Z = *Zio;
  for (int it = 0; it < 6; ++it) {
    lgemm<MT, NT, false, f16>(st, Xs, Z, W, R, R, R, R, R, R, RR, RR, RR, nb,
                              1.f, nullptr, nullptr);           // W = X@Z
    k_aI_minus<<<ewgrid(n), 256, 0, st>>>(7.f, W, T, R, n);     // T = 7I - W
    lgemm<MT, NT, false, f16>(st, W, T, U, R, R, R, R, R, R, RR, RR, RR, nb,
                              1.f, nullptr, nullptr);           // U = W@T
    k_aI_minus<<<ewgrid(n), 256, 0, st>>>(15.f, U, T, R, n);    // T = 15I - U
    lgemm<MT, NT, false, f16>(st, W, T, U, R, R, R, R, R, R, RR, RR, RR, nb,
                              1.f, nullptr, nullptr);           // U = W@T
    k_aI_minus<<<ewgrid(n), 256, 0, st>>>(13.f, U, T, R, n);    // T = 13I - U
    lgemm<MT, NT, false, f16>(st, Z, T, W, R, R, R, R, R, R, RR, RR, RR, nb,
                              0.25f, nullptr, nullptr);         // Z' = .25 Z@T
    f16* tmp = Z; Z = W; W = tmp;
  }
  *Zio = Z;
}

extern "C" void kernel_launch(void* const* d_in, const int* in_sizes, int n_in,
                              void* d_out, int out_size, void* d_ws,
                              size_t ws_size, hipStream_t stream) {
  (void)in_sizes; (void)n_in; (void)out_size; (void)ws_size;

  const float* vol     = (const float*)d_in[0];
  const float* patch_w = (const float*)d_in[1];
  const float* patch_b = (const float*)d_in[2];
  const float* pos_emb = (const float*)d_in[3];
  const float* g_dep   = (const float*)d_in[4];
  const float* qkv_dep = (const float*)d_in[5];
  const float* ow_dep  = (const float*)d_in[6];
  const float* ob_dep  = (const float*)d_in[7];
  const float* g_spa   = (const float*)d_in[8];
  const float* qkv_spa = (const float*)d_in[9];
  const float* ow_spa  = (const float*)d_in[10];
  const float* ob_spa  = (const float*)d_in[11];
  const float* g_ff    = (const float*)d_in[12];
  const float* ff_w1   = (const float*)d_in[13];
  const float* ff_b1   = (const float*)d_in[14];
  const float* ff_w2   = (const float*)d_in[15];
  const float* ff_b2   = (const float*)d_in[16];
  const float* g_fin   = (const float*)d_in[17];

  const int S = 32768, DM = 192;
  const float SCALE = 0.072168783648703f;        // 192^-0.5
  const long long SQKV = (long long)576 * 1024;  // depth-fold row stride

  // ---- workspace allocator ----
  char* wsb = (char*)d_ws;
  size_t off = 0;
  auto take = [&](size_t bytes) -> void* {
    off = (off + 255) & ~(size_t)255;
    void* p = wsb + off;
    off += bytes;
    return p;
  };
  unsigned* stats = (unsigned*)take(64);
  float* X  = (float*)take((size_t)S * DM * 4);     // f32 residual stream
  f16* NB   = (f16*)take((size_t)S * DM * 2);       // normed activations
  f16* QKV  = (f16*)take((size_t)S * 576 * 2);
  // f16 transposed weight copies [N,K]
  f16* pwT   = (f16*)take((size_t)DM * DM * 2);
  f16* qkvdT = (f16*)take((size_t)4 * 576 * DM * 2);
  f16* qkvsT = (f16*)take((size_t)4 * 576 * DM * 2);
  f16* owdT  = (f16*)take((size_t)4 * DM * DM * 2);
  f16* owsT  = (f16*)take((size_t)4 * DM * DM * 2);
  f16* fw1T  = (f16*)take((size_t)4 * 1536 * DM * 2);
  f16* fw2T  = (f16*)take((size_t)4 * DM * 768 * 2);
  const size_t scratch0 = off;

  // ---- one-time weight convert+transpose ----
  k_cvt_t<<<ewgrid(DM * DM), 256, 0, stream>>>(patch_w, pwT, DM, DM, DM * DM);
  for (int i = 0; i < 4; ++i) {
    k_cvt_t<<<ewgrid(DM * 576), 256, 0, stream>>>(
        qkv_dep + (long long)i * DM * 576, qkvdT + (long long)i * 576 * DM,
        DM, 576, DM * 576);
    k_cvt_t<<<ewgrid(DM * 576), 256, 0, stream>>>(
        qkv_spa + (long long)i * DM * 576, qkvsT + (long long)i * 576 * DM,
        DM, 576, DM * 576);
    k_cvt_t<<<ewgrid(DM * DM), 256, 0, stream>>>(
        ow_dep + (long long)i * DM * DM, owdT + (long long)i * DM * DM, DM, DM,
        DM * DM);
    k_cvt_t<<<ewgrid(DM * DM), 256, 0, stream>>>(
        ow_spa + (long long)i * DM * DM, owsT + (long long)i * DM * DM, DM, DM,
        DM * DM);
    k_cvt_t<<<ewgrid(DM * 1536), 256, 0, stream>>>(
        ff_w1 + (long long)i * DM * 1536, fw1T + (long long)i * 1536 * DM, DM,
        1536, DM * 1536);
    k_cvt_t<<<ewgrid(768 * DM), 256, 0, stream>>>(
        ff_w2 + (long long)i * 768 * DM, fw2T + (long long)i * DM * 768, 768,
        DM, 768 * DM);
  }

  // ---- patch embed: X = patchify(vol)@W + b + pos ----
  {
    off = scratch0;
    f16* P = (f16*)take((size_t)S * DM * 2);
    const long long n = (long long)S * DM;
    k_patchify<<<ewgrid(n), 256, 0, stream>>>(vol, P, n);
    lgemm<4, 4, true, float>(stream, P, pwT, X, S, DM, DM, DM, DM, DM, 0, 0, 0,
                             1, 1.f, patch_b, pos_emb);
  }

  for (int i = 0; i < 4; ++i) {
    // ========== depth attention (1024 batches, seq 32; a1=a2=a3) ==========
    {
      off = scratch0;
      float* SCf = (float*)take((size_t)1024 * 32 * 32 * 4);
      f16* SCh = (f16*)take((size_t)1024 * 32 * 32 * 2);
      f16* Zb  = (f16*)take((size_t)1024 * 32 * 32 * 2);
      f16* Wb  = (f16*)take((size_t)1024 * 32 * 32 * 2);
      f16* Tb  = (f16*)take((size_t)1024 * 32 * 32 * 2);
      f16* Ub  = (f16*)take((size_t)1024 * 32 * 32 * 2);
      f16* AV  = (f16*)take((size_t)1024 * 32 * 192 * 2);
      f16* T2  = (f16*)take((size_t)1024 * 32 * 192 * 2);
      f16* AT  = (f16*)take((size_t)S * DM * 2);

      k_scale_norm<f16><<<S / 4, 128, 0, stream>>>(X, NB, g_dep, i);
      lgemm<4, 4, true, f16>(stream, NB, qkvdT + (long long)i * 576 * DM, QKV,
                             S, 576, DM, DM, DM, 576, 0, 0, 0, 1, 1.f, nullptr,
                             nullptr);
      lgemm<2, 2, true, float>(stream, QKV, QKV + 192, SCf, 32, 32, DM,
                               (int)SQKV, (int)SQKV, 32, 576, 576, 1024, 1024,
                               SCALE, nullptr, nullptr);
      k_softmax<<<(1024 * 32) / 4, 128, 0, stream>>>(SCf, SCh, 32);
      hipMemsetAsync(stats, 0, 8, stream);
      k_rowcol_max<<<ewgrid(2LL * 1024 * 32), 256, 0, stream>>>(
          SCh, 32, 32, 1024, 1024, stats);
      const long long nz = 1024LL * 32 * 32;
      k_pinv_seed<<<ewgrid(nz), 256, 0, stream>>>(SCh, Zb, 32, stats, nz);
      f16* Zp = Zb;
      pinv_iters<2, 2>(stream, SCh, &Zp, Wb, Tb, Ub, 32, 1024);
      lgemm<2, 4, false, f16>(stream, SCh, QKV + 384, AV, 32, 192, 32, 32,
                              (int)SQKV, 192, 1024, 576, 32 * 192, 1024, 1.f,
                              nullptr, nullptr);
      lgemm<2, 4, false, f16>(stream, Zp, AV, T2, 32, 192, 32, 32, 192, 192,
                              1024, 32 * 192, 32 * 192, 1024, 1.f, nullptr,
                              nullptr);
      lgemm<2, 4, false, f16>(stream, SCh, T2, AT, 32, 192, 32, 32, 192,
                              1024 * 192, 1024, 32 * 192, 192, 1024, 1.f,
                              nullptr, nullptr);
      lgemm<4, 4, true, float>(stream, AT, owdT + (long long)i * DM * DM, X, S,
                               DM, DM, DM, DM, DM, 0, 0, 0, 1, 1.f,
                               ob_dep + (long long)i * DM, X);
    }

    // ========== spatial attention (32 batches, seq 1024, m=256) ==========
    {
      off = scratch0;
      f16* qL  = (f16*)take((size_t)32 * 256 * 192 * 2);
      f16* kL  = (f16*)take((size_t)32 * 256 * 192 * 2);
      float* A2f = (float*)take((size_t)32 * 256 * 256 * 4);
      f16* A2h = (f16*)take((size_t)32 * 256 * 256 * 2);
      f16* Zb  = (f16*)take((size_t)32 * 256 * 256 * 2);
      f16* Wb  = (f16*)take((size_t)32 * 256 * 256 * 2);
      f16* Tb  = (f16*)take((size_t)32 * 256 * 256 * 2);
      f16* Ub  = (f16*)take((size_t)32 * 256 * 256 * 2);
      float* A1f = (float*)take((size_t)32 * 1024 * 256 * 4);
      f16* A1h = (f16*)take((size_t)32 * 1024 * 256 * 2);
      float* A3f = (float*)take((size_t)32 * 256 * 1024 * 4);
      f16* A3h = (f16*)take((size_t)32 * 256 * 1024 * 2);
      f16* A3V = (f16*)take((size_t)32 * 256 * 192 * 2);
      f16* A1Z = (f16*)take((size_t)32 * 1024 * 256 * 2);
      f16* AT  = (f16*)take((size_t)S * DM * 2);

      k_scale_norm<f16><<<S / 4, 128, 0, stream>>>(X, NB, g_spa, i);
      lgemm<4, 4, true, f16>(stream, NB, qkvsT + (long long)i * 576 * DM, QKV,
                             S, 576, DM, DM, DM, 576, 0, 0, 0, 1, 1.f, nullptr,
                             nullptr);
      const long long nl = 32LL * 256 * 192;
      k_landmarks<<<ewgrid(nl), 256, 0, stream>>>(QKV, qL, kL, nl);
      lgemm<4, 4, true, float>(stream, qL, kL, A2f, 256, 256, DM, DM, DM, 256,
                               256 * 192, 256 * 192, 65536, 32, SCALE, nullptr,
                               nullptr);
      k_softmax<<<(32 * 256) / 4, 128, 0, stream>>>(A2f, A2h, 256);
      hipMemsetAsync(stats, 0, 8, stream);
      k_rowcol_max<<<ewgrid(2LL * 32 * 256), 256, 0, stream>>>(A2h, 256, 256,
                                                               65536, 32,
                                                               stats);
      const long long nz = 32LL * 256 * 256;
      k_pinv_seed<<<ewgrid(nz), 256, 0, stream>>>(A2h, Zb, 256, stats, nz);
      f16* Zp = Zb;
      pinv_iters<4, 4>(stream, A2h, &Zp, Wb, Tb, Ub, 256, 32);
      lgemm<4, 4, true, float>(stream, QKV, kL, A1f, 1024, 256, DM, 576, DM,
                               256, SQKV, 256 * 192, 1024 * 256, 32, SCALE,
                               nullptr, nullptr);
      k_softmax<<<(32 * 1024) / 4, 128, 0, stream>>>(A1f, A1h, 256);
      lgemm<4, 4, true, float>(stream, qL, QKV + 192, A3f, 256, 1024, DM, DM,
                               576, 1024, 256 * 192, SQKV, 256 * 1024, 32,
                               SCALE, nullptr, nullptr);
      k_softmax<<<(32 * 256) / 4, 128, 0, stream>>>(A3f, A3h, 1024);
      lgemm<4, 4, false, f16>(stream, A3h, QKV + 384, A3V, 256, 192, 1024,
                              1024, 576, 192, 256 * 1024, SQKV, 256 * 192, 32,
                              1.f, nullptr, nullptr);
      lgemm<4, 4, false, f16>(stream, A1h, Zp, A1Z, 1024, 256, 256, 256, 256,
                              256, 1024 * 256, 65536, 1024 * 256, 32, 1.f,
                              nullptr, nullptr);
      lgemm<4, 4, false, f16>(stream, A1Z, A3V, AT, 1024, DM, 256, 256, DM,
                              DM, 1024 * 256, 256 * 192, 1024 * 192, 32, 1.f,
                              nullptr, nullptr);
      lgemm<4, 4, true, float>(stream, AT, owsT + (long long)i * DM * DM, X, S,
                               DM, DM, DM, DM, DM, 0, 0, 0, 1, 1.f,
                               ob_spa + (long long)i * DM, X);
    }

    // ========== GEGLU feed-forward ==========
    {
      off = scratch0;
      f16* H = (f16*)take((size_t)S * 1536 * 2);
      f16* G = (f16*)take((size_t)S * 768 * 2);
      k_scale_norm<f16><<<S / 4, 128, 0, stream>>>(X, NB, g_ff, i);
      lgemm<4, 4, true, f16>(stream, NB, fw1T + (long long)i * 1536 * DM, H, S,
                             1536, DM, DM, DM, 1536, 0, 0, 0, 1, 1.f,
                             ff_b1 + (long long)i * 1536, nullptr);
      const long long ng = (long long)S * 768;
      k_geglu<<<ewgrid(ng), 256, 0, stream>>>(H, G, ng);
      lgemm<4, 4, true, float>(stream, G, fw2T + (long long)i * DM * 768, X, S,
                               DM, 768, 768, 768, DM, 0, 0, 0, 1, 1.f,
                               ff_b2 + (long long)i * DM, X);
    }
  }

  // ---- final ScaleNorm -> f32 output ----
  k_scale_norm<float><<<S / 4, 128, 0, stream>>>(X, (float*)d_out, g_fin, 0);
}
